// SPGAT_54820962566700
// MI455X (gfx1250) — matrix operand order repaired and checked by reading, not compiled, not run on hardware
//
#include <hip/hip_runtime.h>
#include <hip/hip_bf16.h>
#include <math.h>

#define N_NODES   50000
#define N_EDGES   1600000
#define E_TOT     (N_EDGES + N_NODES)   // self loops appended logically
#define IN_DIM    128
#define HID       64
#define HEADS     4
#define HC        (HID*HEADS)           // 256
#define N_GRAPHS  64
#define OUT_DIM   10
#define NEG_SLOPE 0.2f

typedef __attribute__((ext_vector_type(16))) __bf16 v16bf;
typedef __attribute__((ext_vector_type(8)))  float  v8f;

// ---------------------------------------------------------------- utilities
__global__ void k_fill(float* __restrict__ p, float v, int n) {
  int i = blockIdx.x * blockDim.x + threadIdx.x;
  if (i < n) p[i] = v;
}

// W[K,N] f32 -> Bt[N,K] bf16 (transposed, for contiguous per-lane K loads)
__global__ void k_w_to_bf16t(const float* __restrict__ W, __bf16* __restrict__ Bt,
                             int K, int N) {
  int i = blockIdx.x * blockDim.x + threadIdx.x; // i = n*K + k
  if (i < K * N) {
    int n = i / K, k = i - n * K;
    Bt[i] = (__bf16)W[k * N + n];
  }
}

// exact float atomic max via sign-split int/uint atomics (init must be -inf)
__device__ __forceinline__ void atomicMaxF(float* addr, float val) {
  if (val >= 0.f) atomicMax((int*)addr, __float_as_int(val));
  else            atomicMin((unsigned int*)addr, __float_as_uint(val));
}

// ---------------------------------------------------------------- WMMA GEMM
// C[M,N] = A[M,K](f32 -> bf16 in-reg) @ Bt[N,K](bf16) (+ optional bias)
// Each wave computes a 16x64 tile (4 independent accumulators -> 4-way WMMA
// ILP, A fragment reused 4x). block = 128 (4 waves, consecutive row tiles);
// grid = (ceil(M/16/4), N/64).
__global__ void k_gemm_bf16(const float* __restrict__ A, const __bf16* __restrict__ Bt,
                            const float* __restrict__ bias, float* __restrict__ C,
                            int M, int N, int K, int rowTiles) {
  const int lane = threadIdx.x & 31;
  const int wave = threadIdx.x >> 5;
  const int rowTile = blockIdx.x * 4 + wave;
  if (rowTile >= rowTiles) return;          // wave-uniform: EXEC all-ones inside
  const int half = lane >> 4;               // which 16-lane half
  const int l16  = lane & 15;
  const int rowBase = rowTile * 16;
  const int colBase = blockIdx.y * 64;

  const float*  Arow = A  + (size_t)(rowBase + l16) * K;
  const __bf16* B0   = Bt + (size_t)(colBase + l16) * K;   // col tile j at +j*16*K

  // A 16x32 layout (matches B): lanes 0-15 hold K 0..15, lanes 16-31 hold K 16..31
  const int koffA0 = half * 16;
  const int koffA1 = half * 16 + 8;
  // B 32x16 layout: lanes 0-15 hold K 0..15, lanes 16-31 hold K 16..31 (contiguous in Bt)
  const int koffB  = half * 16;

  v8f acc0 = {}, acc1 = {}, acc2 = {}, acc3 = {};
  for (int k = 0; k < K; k += 32) {
    __builtin_prefetch(Arow + k + 32, 0, 3);
    float4 a0 = *(const float4*)(Arow + k + koffA0);
    float4 a1 = *(const float4*)(Arow + k + koffA0 + 4);
    float4 a2 = *(const float4*)(Arow + k + koffA1);
    float4 a3 = *(const float4*)(Arow + k + koffA1 + 4);
    v16bf a;
    a[0]  = (__bf16)a0.x; a[1]  = (__bf16)a0.y; a[2]  = (__bf16)a0.z; a[3]  = (__bf16)a0.w;
    a[4]  = (__bf16)a1.x; a[5]  = (__bf16)a1.y; a[6]  = (__bf16)a1.z; a[7]  = (__bf16)a1.w;
    a[8]  = (__bf16)a2.x; a[9]  = (__bf16)a2.y; a[10] = (__bf16)a2.z; a[11] = (__bf16)a2.w;
    a[12] = (__bf16)a3.x; a[13] = (__bf16)a3.y; a[14] = (__bf16)a3.z; a[15] = (__bf16)a3.w;
    v16bf b0 = *(const v16bf*)(B0 + 0 * 16 * K + k + koffB);
    v16bf b1 = *(const v16bf*)(B0 + 1 * 16 * K + k + koffB);
    v16bf b2 = *(const v16bf*)(B0 + 2 * 16 * K + k + koffB);
    v16bf b3 = *(const v16bf*)(B0 + 3 * 16 * K + k + koffB);
    acc0 = __builtin_amdgcn_wmma_f32_16x16x32_bf16(false, a, false, b0, (short)0, acc0, false, false);
    acc1 = __builtin_amdgcn_wmma_f32_16x16x32_bf16(false, a, false, b1, (short)0, acc1, false, false);
    acc2 = __builtin_amdgcn_wmma_f32_16x16x32_bf16(false, a, false, b2, (short)0, acc2, false, false);
    acc3 = __builtin_amdgcn_wmma_f32_16x16x32_bf16(false, a, false, b3, (short)0, acc3, false, false);
  }
  // C/D layout: VGPR i -> row = rowBase + half*8 + i, col = colTile*16 + l16
#pragma unroll
  for (int j = 0; j < 4; ++j) {
    const v8f acc = (j == 0) ? acc0 : (j == 1) ? acc1 : (j == 2) ? acc2 : acc3;
    const int c = colBase + j * 16 + l16;
    const float bv = bias ? bias[c] : 0.f;
#pragma unroll
    for (int i = 0; i < 8; ++i) {
      C[(size_t)(rowBase + half * 8 + i) * N + c] = acc[i] + bv;
    }
  }
}

// ---------------------------------------------------------------- GAT edge path
// per-node per-head attention scores: es/ed[n,h] = <h[n,h,:], a_src/dst[h,:]>
__global__ void k_scores(const float* __restrict__ h, const float* __restrict__ asrc,
                         const float* __restrict__ adst,
                         float* __restrict__ es, float* __restrict__ ed) {
  int i = blockIdx.x * blockDim.x + threadIdx.x;   // n*HEADS + hd
  if (i >= N_NODES * HEADS) return;
  int n = i >> 2, hd = i & 3;
  const float* hp = h + (size_t)n * HC + hd * HID;
  const float* as = asrc + hd * HID;
  const float* ad = adst + hd * HID;
  float s0 = 0.f, s1 = 0.f;
  for (int c = 0; c < HID; ++c) { float v = hp[c]; s0 += v * as[c]; s1 += v * ad[c]; }
  es[i] = s0; ed[i] = s1;
}

__device__ __forceinline__ void edge_sd(const int* __restrict__ ei, int e, int& s, int& d) {
  if (e < N_EDGES) { s = ei[e]; d = ei[N_EDGES + e]; }
  else             { s = d = e - N_EDGES; }       // self loop
}

// pass 1: e = leaky_relu(es[src]+ed[dst]); segment max over dst
__global__ void k_edge_max(const int* __restrict__ ei, const float* __restrict__ es,
                           const float* __restrict__ ed, float* __restrict__ ebuf,
                           float* __restrict__ m) {
  int i = blockIdx.x * blockDim.x + threadIdx.x;   // e*HEADS + hd
  if (i >= E_TOT * HEADS) return;
  int e = i >> 2, hd = i & 3, s, d;
  edge_sd(ei, e, s, d);
  float v = es[s * HEADS + hd] + ed[d * HEADS + hd];
  v = v > 0.f ? v : NEG_SLOPE * v;
  ebuf[i] = v;
  atomicMaxF(&m[d * HEADS + hd], v);
}

// pass 2: ex = exp(e - m[dst]); segment sum over dst
__global__ void k_edge_exp(const int* __restrict__ ei, float* __restrict__ ebuf,
                           const float* __restrict__ m, float* __restrict__ ssum) {
  int i = blockIdx.x * blockDim.x + threadIdx.x;
  if (i >= E_TOT * HEADS) return;
  int e = i >> 2, hd = i & 3, s, d;
  edge_sd(ei, e, s, d);
  float mm = m[d * HEADS + hd];
  if (!(mm > -3.0e38f && mm < 3.0e38f)) mm = 0.f;  // mirror jnp.where(isfinite)
  float ex = expf(ebuf[i] - mm);
  ebuf[i] = ex;
  atomicAdd(&ssum[d * HEADS + hd], ex);
}

// pass 3: out[dst] += h[src] * alpha  (wave per edge, 8 channels per lane)
__global__ void k_edge_aggr(const int* __restrict__ ei, const float* __restrict__ h,
                            const float* __restrict__ ebuf, const float* __restrict__ ssum,
                            float* __restrict__ out) {
  int gt = blockIdx.x * blockDim.x + threadIdx.x;
  int e = gt >> 5;
  if (e >= E_TOT) return;
  int lane = gt & 31;
  int s, d;  edge_sd(ei, e, s, d);
  int hd = lane >> 3;                                // 8 lanes per head (64 ch / 8)
  float alpha = ebuf[e * HEADS + hd] / (ssum[d * HEADS + hd] + 1e-16f);
  const float4* hp = (const float4*)(h + (size_t)s * HC + lane * 8);
  float4 v0 = hp[0], v1 = hp[1];
  float* op = out + (size_t)d * HC + lane * 8;
  atomicAdd(op + 0, v0.x * alpha); atomicAdd(op + 1, v0.y * alpha);
  atomicAdd(op + 2, v0.z * alpha); atomicAdd(op + 3, v0.w * alpha);
  atomicAdd(op + 4, v1.x * alpha); atomicAdd(op + 5, v1.y * alpha);
  atomicAdd(op + 6, v1.z * alpha); atomicAdd(op + 7, v1.w * alpha);
}

// x = elu(out + b) over [N, HC]
__global__ void k_bias_elu(const float* __restrict__ o, const float* __restrict__ b,
                           float* __restrict__ x) {
  int i = blockIdx.x * blockDim.x + threadIdx.x;
  if (i >= N_NODES * HC) return;
  float v = o[i] + b[i & (HC - 1)];
  x[i] = v > 0.f ? v : (expf(v) - 1.f);
}

// last layer: mean over heads + b2  -> [N, HID]
__global__ void k_mean_heads(const float* __restrict__ o, const float* __restrict__ b2,
                             float* __restrict__ x) {
  int i = blockIdx.x * blockDim.x + threadIdx.x;
  if (i >= N_NODES * HID) return;
  int n = i >> 6, c = i & 63;
  const float* p = o + (size_t)n * HC + c;
  x[i] = 0.25f * (p[0] + p[HID] + p[2 * HID] + p[3 * HID]) + b2[c];
}

__global__ void k_pool(const float* __restrict__ x, const int* __restrict__ batch,
                       float* __restrict__ pooled) {
  int i = blockIdx.x * blockDim.x + threadIdx.x;
  if (i >= N_NODES * HID) return;
  int n = i >> 6, c = i & 63;
  atomicAdd(&pooled[batch[n] * HID + c], x[i]);
}

__global__ void k_count(const int* __restrict__ batch, float* __restrict__ cnt) {
  int n = blockIdx.x * blockDim.x + threadIdx.x;
  if (n < N_NODES) atomicAdd(&cnt[batch[n]], 1.f);
}

// per-graph MLP: block per graph, 64 threads
__global__ void k_mlp(const float* __restrict__ pooled, const float* __restrict__ cnt,
                      const float* __restrict__ mW1, const float* __restrict__ mb1,
                      const float* __restrict__ mW2, const float* __restrict__ mb2,
                      float* __restrict__ out) {
  __shared__ float p[HID], hm[HID];
  int g = blockIdx.x, t = threadIdx.x;
  float inv = 1.f / fmaxf(cnt[g], 1.f);
  p[t] = pooled[g * HID + t] * inv;
  __syncthreads();
  float acc = mb1[t];
  for (int c = 0; c < HID; ++c) acc += p[c] * mW1[c * HID + t];
  hm[t] = fmaxf(acc, 0.f);
  __syncthreads();
  if (t < OUT_DIM) {
    float o = mb2[t];
    for (int c = 0; c < HID; ++c) o += hm[c] * mW2[c * OUT_DIM + t];
    out[g * OUT_DIM + t] = o;
  }
}

// ---------------------------------------------------------------- launch
extern "C" void kernel_launch(void* const* d_in, const int* in_sizes, int n_in,
                              void* d_out, int out_size, void* d_ws, size_t ws_size,
                              hipStream_t stream) {
  (void)in_sizes; (void)n_in; (void)out_size; (void)ws_size;
  const float* imgs  = (const float*)d_in[0];
  const int*   ei    = (const int*)  d_in[1];
  const int*   batch = (const int*)  d_in[2];
  const float* W_agg = (const float*)d_in[3];  const float* b_agg = (const float*)d_in[4];
  const float* W0 = (const float*)d_in[5];  const float* as0 = (const float*)d_in[6];
  const float* ad0 = (const float*)d_in[7]; const float* b0  = (const float*)d_in[8];
  const float* W1 = (const float*)d_in[9];  const float* as1 = (const float*)d_in[10];
  const float* ad1 = (const float*)d_in[11];const float* b1  = (const float*)d_in[12];
  const float* W2 = (const float*)d_in[13]; const float* as2 = (const float*)d_in[14];
  const float* ad2 = (const float*)d_in[15];const float* b2  = (const float*)d_in[16];
  const float* mW1 = (const float*)d_in[17];const float* mb1 = (const float*)d_in[18];
  const float* mW2 = (const float*)d_in[19];const float* mb2 = (const float*)d_in[20];
  float* out = (float*)d_out;

  char* ws = (char*)d_ws; size_t off = 0;
  auto alloc = [&](size_t bytes) -> char* {
    char* p = ws + off; off = (off + bytes + 255) & ~(size_t)255; return p;
  };
  float*  bufX   = (float*) alloc((size_t)N_NODES * HC * 4);     // activations
  float*  bufH   = (float*) alloc((size_t)N_NODES * HC * 4);     // h = x@W
  float*  bufO   = (float*) alloc((size_t)N_NODES * HC * 4);     // aggregated
  float*  es     = (float*) alloc((size_t)N_NODES * HEADS * 4);
  float*  ed     = (float*) alloc((size_t)N_NODES * HEADS * 4);
  float*  mbuf   = (float*) alloc((size_t)N_NODES * HEADS * 4);
  float*  ssum   = (float*) alloc((size_t)N_NODES * HEADS * 4);
  float*  ebuf   = (float*) alloc((size_t)E_TOT   * HEADS * 4);
  __bf16* Bt     = (__bf16*)alloc((size_t)HC * HC * 2);
  float*  pooled = (float*) alloc((size_t)N_GRAPHS * HID * 4);
  float*  cnt    = (float*) alloc((size_t)N_GRAPHS * 4);

  const int BLK = 256;
  auto cdiv = [](int a, int b) { return (a + b - 1) / b; };

  auto gemm = [&](const float* A, const float* W, const float* bias, float* C,
                  int M, int N, int K) {
    k_w_to_bf16t<<<cdiv(K * N, BLK), BLK, 0, stream>>>(W, Bt, K, N);
    int rowTiles = M / 16;
    dim3 grid(cdiv(rowTiles, 4), N / 64);
    k_gemm_bf16<<<grid, 128, 0, stream>>>(A, Bt, bias, C, M, N, K, rowTiles);
  };

  auto gat = [&](const float* xin, const float* W, const float* as_, const float* ad_,
                 int K) {
    gemm(xin, W, nullptr, bufH, N_NODES, HC, K);
    k_scores<<<cdiv(N_NODES * HEADS, BLK), BLK, 0, stream>>>(bufH, as_, ad_, es, ed);
    k_fill<<<cdiv(N_NODES * HEADS, BLK), BLK, 0, stream>>>(mbuf, -INFINITY, N_NODES * HEADS);
    k_fill<<<cdiv(N_NODES * HEADS, BLK), BLK, 0, stream>>>(ssum, 0.f, N_NODES * HEADS);
    k_fill<<<cdiv(N_NODES * HC, BLK), BLK, 0, stream>>>(bufO, 0.f, N_NODES * HC);
    k_edge_max<<<cdiv(E_TOT * HEADS, BLK), BLK, 0, stream>>>(ei, es, ed, ebuf, mbuf);
    k_edge_exp<<<cdiv(E_TOT * HEADS, BLK), BLK, 0, stream>>>(ei, ebuf, mbuf, ssum);
    k_edge_aggr<<<cdiv(E_TOT * 32, BLK), BLK, 0, stream>>>(ei, bufH, ebuf, ssum, bufO);
  };

  // aggregator: x = imgs @ W_agg + b_agg   [N, 64]
  gemm(imgs, W_agg, b_agg, bufX, N_NODES, HID, IN_DIM);
  // GAT 0 (K=64) -> elu(out + b0)          [N, 256]
  gat(bufX, W0, as0, ad0, HID);
  k_bias_elu<<<cdiv(N_NODES * HC, BLK), BLK, 0, stream>>>(bufO, b0, bufX);
  // GAT 1 (K=256) -> elu(out + b1)         [N, 256]
  gat(bufX, W1, as1, ad1, HC);
  k_bias_elu<<<cdiv(N_NODES * HC, BLK), BLK, 0, stream>>>(bufO, b1, bufX);
  // GAT 2 (K=256) -> mean heads + b2       [N, 64]
  gat(bufX, W2, as2, ad2, HC);
  float* xfin = bufX;  // reuse
  k_mean_heads<<<cdiv(N_NODES * HID, BLK), BLK, 0, stream>>>(bufO, b2, xfin);
  // global mean pool + MLP
  k_fill<<<cdiv(N_GRAPHS * HID, BLK), BLK, 0, stream>>>(pooled, 0.f, N_GRAPHS * HID);
  k_fill<<<1, BLK, 0, stream>>>(cnt, 0.f, N_GRAPHS);
  k_pool<<<cdiv(N_NODES * HID, BLK), BLK, 0, stream>>>(xfin, batch, pooled);
  k_count<<<cdiv(N_NODES, BLK), BLK, 0, stream>>>(batch, cnt);
  k_mlp<<<N_GRAPHS, HID, 0, stream>>>(pooled, cnt, mW1, mb1, mW2, mb2, out);
}
